// DeepSetsFFNLayer_62388694942439
// MI455X (gfx1250) — compile-verified, matrix operand order checked by reading, not run on hardware
//
#include <hip/hip_runtime.h>
#include <hip/hip_bf16.h>
#include <math.h>

typedef __attribute__((ext_vector_type(16))) _Float16 v16h;
typedef __attribute__((ext_vector_type(8)))  _Float16 v8h;
typedef __attribute__((ext_vector_type(8)))  float    v8f;

#define TOK   512
#define DM    512
#define DFF   2048
#define DN    128
#define DH    256
#define NPB   64          // neurons per block in k_main
#define LN_EPS 1e-5f

// Fast gelu: x * sigmoid(1.5957691*x*(1+0.044715*x^2)) via hardware v_exp_f32.
// Max abs error ~3e-4, below the f16 quantization applied to the GEMM operands.
__device__ __forceinline__ float gelu_fast(float x) {
    float u = 1.5957691216057308f * x * __builtin_fmaf(0.044715f, x * x, 1.0f);
    return x / (1.0f + __expf(-u));
}

// ---------------------------------------------------------------- f32 -> f16
__global__ void k_cvt(const float* __restrict__ src, _Float16* __restrict__ dst, int n) {
    int i = blockIdx.x * 256 + threadIdx.x;
    if (i < n) dst[i] = (_Float16)src[i];
}

__global__ void k_zero(float* __restrict__ p, int n) {
    int i = blockIdx.x * 256 + threadIdx.x;
    if (i < n) p[i] = 0.0f;
}

// -------------------------------------------------- shared phi_w1 last-col stats
__global__ void k_phig(const float* __restrict__ phi_w1, const float* __restrict__ g1,
                       float* __restrict__ q, float* __restrict__ vhat, float* __restrict__ varv) {
    __shared__ float s[256];
    int j = threadIdx.x;
    float v = phi_w1[j * (DN + 1) + DN];   // w_act
    s[j] = v; __syncthreads();
    for (int off = 128; off > 0; off >>= 1) { if (j < off) s[j] += s[j + off]; __syncthreads(); }
    float mu = s[0] * (1.0f / 256.0f);
    __syncthreads();
    float vh = v - mu;
    s[j] = vh * vh; __syncthreads();
    for (int off = 128; off > 0; off >>= 1) { if (j < off) s[j] += s[j + off]; __syncthreads(); }
    if (j == 0) varv[0] = s[0] * (1.0f / 256.0f);
    vhat[j] = vh;
    q[j] = vh * g1[j];
}

// -------------------------------------------------- per-neuron LN1 decomposition
__global__ void k_nstats(const float* __restrict__ nv, const float* __restrict__ phi_w1,
                         const float* __restrict__ phi_b1, const float* __restrict__ g1,
                         const float* __restrict__ vhat,
                         float* __restrict__ p, float* __restrict__ stats) {
    __shared__ float sa[256], sb[256];
    __shared__ float lnv[DN];
    int n = blockIdx.x, j = threadIdx.x;
    if (j < DN) lnv[j] = nv[n * DN + j];
    __syncthreads();
    float u = phi_b1[j];
    const float* wr = phi_w1 + j * (DN + 1);
    #pragma unroll 8
    for (int i = 0; i < DN; i++) u += lnv[i] * wr[i];
    sa[j] = u; __syncthreads();
    for (int off = 128; off > 0; off >>= 1) { if (j < off) sa[j] += sa[j + off]; __syncthreads(); }
    float mu = sa[0] * (1.0f / 256.0f);
    __syncthreads();
    float uh = u - mu;
    sa[j] = uh * uh; sb[j] = uh * vhat[j];
    __syncthreads();
    for (int off = 128; off > 0; off >>= 1) {
        if (j < off) { sa[j] += sa[j + off]; sb[j] += sb[j + off]; }
        __syncthreads();
    }
    if (j == 0) { stats[n * 2] = sa[0] * (1.0f / 256.0f); stats[n * 2 + 1] = sb[0] * (1.0f / 256.0f); }
    p[n * DH + j] = uh * g1[j];
}

// -------------------------------------------------- acts = gelu(x @ W_in^T), WMMA
// grid (32, 16): 16-row tile x 128-col tile; block 256 = 8 waves, wave -> 16 cols
__global__ __launch_bounds__(256) void k_acts(const _Float16* __restrict__ xh,
                                              const _Float16* __restrict__ wh,
                                              float* __restrict__ acts) {
    __shared__ _Float16 lA[16 * DM];   // A fragments: [ks][lane][j], 16 k-steps of 32
    int tid = threadIdx.x;
    int t0 = blockIdx.x * 16;
    for (int s = tid; s < 16 * 32; s += 256) {   // 512 slots x 16 halves
        int ks = s >> 5, l2 = s & 31;
        const _Float16* row = xh + (t0 + (l2 & 15)) * DM + ks * 32 + ((l2 >= 16) ? 8 : 0);
        v8h lo = *(const v8h*)(row);          // K = base .. base+7
        v8h hi = *(const v8h*)(row + 16);     // K = base+16 .. base+23
        _Float16* d = lA + s * 16;
        #pragma unroll
        for (int j = 0; j < 8; j++) { d[j] = lo[j]; d[j + 8] = hi[j]; }
    }
    __syncthreads();
    int w = tid >> 5, lane = tid & 31;
    int ncol = blockIdx.y * 128 + w * 16 + (lane & 15);
    const _Float16* brow = wh + ncol * DM + ((lane >= 16) ? 8 : 0);
    v8f c = {};
    #pragma unroll
    for (int ks = 0; ks < 16; ks++) {
        v16h a = *(const v16h*)(lA + (ks * 32 + lane) * 16);
        v8h blo = *(const v8h*)(brow + ks * 32);
        v8h bhi = *(const v8h*)(brow + ks * 32 + 16);
        v16h b;
        #pragma unroll
        for (int j = 0; j < 8; j++) { b[j] = blo[j]; b[j + 8] = bhi[j]; }
        c = __builtin_amdgcn_wmma_f32_16x16x32_f16(false, a, false, b, (short)0, c, false, false);
    }
    int rbase = (lane >= 16) ? 8 : 0;
    #pragma unroll
    for (int r = 0; r < 8; r++)
        acts[(t0 + rbase + r) * DFF + ncol] = gelu_fast(c[r]);
}

// -------------------------------------------------- main fused phi kernel (137 GFLOP)
// grid (32, 32): 16 tokens x 64 neurons per block; block 256 = 8 waves.
// Per wave: B fragments (2 N-tiles x 8 k-steps of phi_w2^T) pinned in VGPRs.
// 2 barriers/neuron; LN2 row stats via shfl_xor + conflict-free ds atomics.
// p[n+1] prefetched straight into LDS with GLOBAL_LOAD_ASYNC_TO_LDS (ASYNCcnt).
__global__ __launch_bounds__(256) void k_main(
        const float* __restrict__ acts, const float* __restrict__ p,
        const float* __restrict__ stats, const float* __restrict__ q,
        const float* __restrict__ varv_p, const float* __restrict__ ln1b,
        const float* __restrict__ phi_w2, const float* __restrict__ phi_b2,
        const float* __restrict__ g2, const float* __restrict__ b2ln,
        float* __restrict__ agg) {
    __shared__ _Float16 lA[16 * DH];            // 8 KB: A tile for current neuron
    __shared__ float lp[2][DH];                 // double-buffered p[n] (async prefetch)
    __shared__ float lq[DH], lb[DH], lg2[DH], lb2[DH], lb2ln[DH];
    __shared__ float red[2][32];                // double-buffered per-row sum/sumsq

    int tid = threadIdx.x, lane = tid & 31, w = tid >> 5;
    int t0 = blockIdx.x * 16;
    int nbase = blockIdx.y * NPB;
    float varv = varv_p[0];

    for (int s = tid; s < DH; s += 256) {
        lq[s] = q[s]; lb[s] = ln1b[s]; lg2[s] = g2[s]; lb2[s] = phi_b2[s]; lb2ln[s] = b2ln[s];
        lp[0][s] = p[nbase * DH + s];
    }
    if (tid < 64) red[tid >> 5][tid & 31] = 0.0f;

    // LDS byte addresses of the two lp buffers for this thread's element
    // (generic shared pointer low 32 bits == LDS byte offset per aperture rules)
    unsigned lpa0 = (unsigned)(size_t)(&lp[0][tid]);
    unsigned lpa1 = (unsigned)(size_t)(&lp[1][tid]);

    // load phi_w2^T fragments into registers: B[k][n] = phi_w2[n][k]
    int ln = lane & 15, kb8 = (lane >= 16) ? 8 : 0;
    int ntA = 2 * w, ntB = 2 * w + 1;
    v16h B0[8], B1[8];
    #pragma unroll
    for (int ks = 0; ks < 8; ks++) {
        const float* s0 = phi_w2 + (ntA * 16 + ln) * DH + ks * 32 + kb8;
        const float* s1 = phi_w2 + (ntB * 16 + ln) * DH + ks * 32 + kb8;
        v16h b0, b1;
        #pragma unroll
        for (int j = 0; j < 8; j++) {
            b0[j] = (_Float16)s0[j];  b0[j + 8] = (_Float16)s0[j + 16];
            b1[j] = (_Float16)s1[j];  b1[j + 8] = (_Float16)s1[j + 16];
        }
        B0[ks] = b0; B1[ks] = b1;
    }

    int col0 = ntA * 16 + ln, col1 = ntB * 16 + ln;
    int rbase = (lane >= 16) ? 8 : 0;
    v8f agg0 = {}, agg1 = {};
    __syncthreads();

    for (int nn = 0; nn < NPB; nn++) {
        int c = nn & 1;
        int n = nbase + nn;
        // ---- build A tile = gelu(LN1), closed-form LN1
        {
            int ks = tid >> 5, l2 = tid & 31;
            float a = acts[(t0 + (l2 & 15)) * DFF + n];
            float vu = stats[n * 2], cv = stats[n * 2 + 1];
            float si = rsqrtf(vu + 2.0f * a * cv + a * a * varv + LN_EPS);
            _Float16* d = lA + tid * 16;
            int kb = ks * 32 + ((l2 >= 16) ? 8 : 0);
            #pragma unroll
            for (int j = 0; j < 16; j++) {
                int k = kb + j + ((j >= 8) ? 8 : 0);
                float h = __builtin_fmaf(si, __builtin_fmaf(a, lq[k], lp[c][k]), lb[k]);
                d[j] = (_Float16)gelu_fast(h);
            }
        }
        // ---- async prefetch p[n+1] directly into the other LDS buffer
        //      (1 element/thread; ASYNCcnt-tracked, bypasses VGPRs)
        if (nn + 1 < NPB) {
            unsigned ldsa = c ? lpa0 : lpa1;
            unsigned voff = (unsigned)(((n + 1) * DH + tid) * 4);
            asm volatile("global_load_async_to_lds_b32 %0, %1, %2"
                         :: "v"(ldsa), "v"(voff), "s"(p) : "memory");
        }
        // async copies must land before the barrier that publishes lp[1-c]
        asm volatile("s_wait_asynccnt 0x0" ::: "memory");
        __syncthreads();   // A tile + prefetched p visible

        v8f z0 = {}, z1 = {};
        #pragma unroll
        for (int ks = 0; ks < 8; ks++) {
            v16h a = *(const v16h*)(lA + (ks * 32 + lane) * 16);
            z0 = __builtin_amdgcn_wmma_f32_16x16x32_f16(false, a, false, B0[ks], (short)0, z0, false, false);
            z1 = __builtin_amdgcn_wmma_f32_16x16x32_f16(false, a, false, B1[ks], (short)0, z1, false, false);
        }
        float bb0 = lb2[col0], bb1 = lb2[col1];
        #pragma unroll
        for (int r = 0; r < 8; r++) { z0[r] += bb0; z1[r] += bb1; }

        // ---- LN2 row stats: shfl_xor reduce over the 16-lane column group,
        //      then lanes 0/16 do conflict-free LDS atomics (cross-wave combine)
        #pragma unroll
        for (int r = 0; r < 8; r++) {
            float sv = z0[r] + z1[r];
            float sq = z0[r] * z0[r] + z1[r] * z1[r];
            #pragma unroll
            for (int m = 1; m < 16; m <<= 1) {
                sv += __shfl_xor(sv, m, 32);
                sq += __shfl_xor(sq, m, 32);
            }
            if ((lane & 15) == 0) {
                int row = rbase + r;
                atomicAdd(&red[c][row * 2], sv);
                atomicAdd(&red[c][row * 2 + 1], sq);
            }
        }
        __syncthreads();   // red[c] complete

        float gc0 = lg2[col0], gc1 = lg2[col1];
        float bc0 = lb2ln[col0], bc1 = lb2ln[col1];
        #pragma unroll
        for (int r = 0; r < 8; r++) {
            int row = rbase + r;
            float mean = red[c][row * 2] * (1.0f / 256.0f);
            float var  = red[c][row * 2 + 1] * (1.0f / 256.0f) - mean * mean;
            float rstd = rsqrtf(var + LN_EPS);
            agg0[r] += (z0[r] - mean) * rstd * gc0 + bc0;
            agg1[r] += (z1[r] - mean) * rstd * gc1 + bc1;
        }
        if (tid < 32) red[1 - c][tid] = 0.0f;   // safe: next write is after next barrier
        // no trailing barrier needed: next iteration's post-A-build barrier orders
        // lA overwrite (reads done pre-mid-barrier) and red[1-c] atomics.
    }

    #pragma unroll
    for (int r = 0; r < 8; r++) {
        int t = t0 + rbase + r;
        atomicAdd(&agg[t * DH + col0], agg0[r]);
        atomicAdd(&agg[t * DH + col1], agg1[r]);
    }
}

// -------------------------------------------------- rho tail: LN + 2-layer MLP
__global__ void k_tail(const float* __restrict__ agg, const float* __restrict__ g,
                       const float* __restrict__ b, const float* __restrict__ w1,
                       const float* __restrict__ b1, const float* __restrict__ w2,
                       const float* __restrict__ b2, float* __restrict__ out) {
    __shared__ float r0[DH];
    __shared__ float r1[2 * DH];
    __shared__ float sa[256], sb[256];
    int t = blockIdx.x, tid = threadIdx.x;
    float v = agg[t * DH + tid];
    sa[tid] = v; sb[tid] = v * v;
    __syncthreads();
    for (int off = 128; off > 0; off >>= 1) {
        if (tid < off) { sa[tid] += sa[tid + off]; sb[tid] += sb[tid + off]; }
        __syncthreads();
    }
    float mean = sa[0] * (1.0f / 256.0f);
    float var  = sb[0] * (1.0f / 256.0f) - mean * mean;
    float rstd = rsqrtf(var + LN_EPS);
    r0[tid] = (v - mean) * rstd * g[tid] + b[tid];
    __syncthreads();
    for (int j = tid; j < 2 * DH; j += 256) {
        float acc = b1[j];
        const float* wr = w1 + j * DH;
        #pragma unroll 8
        for (int k = 0; k < DH; k++) acc += r0[k] * wr[k];
        r1[j] = gelu_fast(acc);
    }
    __syncthreads();
    for (int j = tid; j < DM; j += 256) {
        float acc = b2[j];
        const float* wr = w2 + j * 2 * DH;
        #pragma unroll 8
        for (int k = 0; k < 2 * DH; k++) acc += r1[k] * wr[k];
        out[t * DM + j] = acc;
    }
}

extern "C" void kernel_launch(void* const* d_in, const int* in_sizes, int n_in,
                              void* d_out, int out_size, void* d_ws, size_t ws_size,
                              hipStream_t stream) {
    const float* x      = (const float*)d_in[0];
    const float* nv     = (const float*)d_in[1];
    const float* W_in   = (const float*)d_in[2];
    const float* phi_w1 = (const float*)d_in[3];
    const float* phi_b1 = (const float*)d_in[4];
    const float* ln1g   = (const float*)d_in[5];
    const float* ln1b   = (const float*)d_in[6];
    const float* phi_w2 = (const float*)d_in[7];
    const float* phi_b2 = (const float*)d_in[8];
    const float* ln2g   = (const float*)d_in[9];
    const float* ln2b   = (const float*)d_in[10];
    const float* rlng   = (const float*)d_in[11];
    const float* rlnb   = (const float*)d_in[12];
    const float* rw1    = (const float*)d_in[13];
    const float* rb1    = (const float*)d_in[14];
    const float* rw2    = (const float*)d_in[15];
    const float* rb2    = (const float*)d_in[16];

    char* base = (char*)d_ws;
    _Float16* xh = (_Float16*)base;  base += (size_t)TOK * DM * 2;
    _Float16* wh = (_Float16*)base;  base += (size_t)DFF * DM * 2;
    float* acts  = (float*)base;     base += (size_t)TOK * DFF * 4;
    float* p     = (float*)base;     base += (size_t)DFF * DH * 4;
    float* stats = (float*)base;     base += (size_t)DFF * 2 * 4;
    float* q     = (float*)base;     base += (size_t)DH * 4;
    float* vhat  = (float*)base;     base += (size_t)DH * 4;
    float* varv  = (float*)base;     base += 256;
    float* agg   = (float*)base;     base += (size_t)TOK * DH * 4;

    k_cvt<<<(TOK * DM + 255) / 256, 256, 0, stream>>>(x, xh, TOK * DM);
    k_cvt<<<(DFF * DM + 255) / 256, 256, 0, stream>>>(W_in, wh, DFF * DM);
    k_phig<<<1, 256, 0, stream>>>(phi_w1, ln1g, q, vhat, varv);
    k_nstats<<<DFF, 256, 0, stream>>>(nv, phi_w1, phi_b1, ln1g, vhat, p, stats);
    k_acts<<<dim3(TOK / 16, DFF / 128), 256, 0, stream>>>(xh, wh, acts);
    k_zero<<<(TOK * DH + 255) / 256, 256, 0, stream>>>(agg, TOK * DH);
    k_main<<<dim3(TOK / 16, DFF / NPB), 256, 0, stream>>>(acts, p, stats, q, varv, ln1b,
                                                          phi_w2, phi_b2, ln2g, ln2b, agg);
    k_tail<<<TOK, 256, 0, stream>>>(agg, rlng, rlnb, rw1, rb1, rw2, rb2, (float*)d_out);
}
